// GraphTransformerScoreModel_23029614641881
// MI455X (gfx1250) — compile-verified, matrix operand order checked by reading, not run on hardware
//
#include <hip/hip_runtime.h>
#include <hip/hip_bf16.h>

// ---------------------------------------------------------------------------
// GraphTransformerScoreModel forward for gfx1250 (MI455X, wave32, WMMA).
// Dense GEMMs: v_wmma_f32_16x16x32_f16, f32 accumulate, unguarded b128 loads.
// Weights are pre-transposed/converted to f16 (NxKp) so B fragments are two
// contiguous 16B loads; A fragments are four b128 f32 loads + pack-convert.
// ---------------------------------------------------------------------------

typedef __attribute__((ext_vector_type(16))) _Float16 v16h;
typedef __attribute__((ext_vector_type(8)))  _Float16 v8h;
typedef __attribute__((ext_vector_type(8)))  float    v8f;
typedef __attribute__((ext_vector_type(4)))  float    v4f;

#define HID 128
#define NL 5

__device__ __forceinline__ float silu_f(float x) { return x / (1.0f + __expf(-x)); }

// ---------------------------------------------------------------------------
// WMMA GEMM: C[M,N] = act(A[M,K] * B[K,N] + bias)
//   A : f32 row-major, stride lda (floats)
//   Bt: f16, transposed N x ldb (ldb >= K, zero-padded), row-major
//   C : f32 or f16, stride ldc
// One wave computes a 16x32 output region (2 WMMA tiles, shared A fragment).
// Requires: K % 32 == 0, M % 16 == 0, N % 32 == 0 (true at all call sites).
// ---------------------------------------------------------------------------
template<int ACT, bool OUT_F16>
__global__ void wmma_gemm(const float* __restrict__ A, int lda,
                          const _Float16* __restrict__ Bt, int ldb,
                          const float* __restrict__ bias,
                          void* __restrict__ Cout, int ldc,
                          int M, int N, int K)
{
    const int wavesPerBlock = blockDim.x >> 5;
    const int wid    = blockIdx.x * wavesPerBlock + (threadIdx.x >> 5);
    const int tilesN = N >> 5;                 // 32-wide column strips
    const int tiles  = (M >> 4) * tilesN;
    if (wid >= tiles) return;                  // uniform whole-wave exit

    const int tm   = wid / tilesN;
    const int tn   = wid % tilesN;
    const int lane = threadIdx.x & 31;
    const int l15  = lane & 15;
    const int koff = (lane & 16) ? 8 : 0;      // ISA 16-bit A/B 16x32 lane layout

    const float*    Arow = A  + (size_t)(tm * 16 + l15) * lda + koff;
    const _Float16* B0   = Bt + (size_t)(tn * 32 + l15) * ldb + koff;
    const _Float16* B1   = B0 + (size_t)16 * ldb;

    v8f acc0 = {}, acc1 = {};
    for (int kk = 0; kk < K; kk += 32) {
        // A fragment: rows of 8 contiguous floats at K = kk+koff.. and kk+16+koff..
        v4f a0 = *(const v4f*)(Arow + kk);
        v4f a1 = *(const v4f*)(Arow + kk + 4);
        v4f a2 = *(const v4f*)(Arow + kk + 16);
        v4f a3 = *(const v4f*)(Arow + kk + 20);
        v16h a;
        a[0]  = (_Float16)a0.x; a[1]  = (_Float16)a0.y; a[2]  = (_Float16)a0.z; a[3]  = (_Float16)a0.w;
        a[4]  = (_Float16)a1.x; a[5]  = (_Float16)a1.y; a[6]  = (_Float16)a1.z; a[7]  = (_Float16)a1.w;
        a[8]  = (_Float16)a2.x; a[9]  = (_Float16)a2.y; a[10] = (_Float16)a2.z; a[11] = (_Float16)a2.w;
        a[12] = (_Float16)a3.x; a[13] = (_Float16)a3.y; a[14] = (_Float16)a3.z; a[15] = (_Float16)a3.w;

        // B fragments: contiguous f16, two 16B loads each
        v8h b0lo = *(const v8h*)(B0 + kk);
        v8h b0hi = *(const v8h*)(B0 + kk + 16);
        v8h b1lo = *(const v8h*)(B1 + kk);
        v8h b1hi = *(const v8h*)(B1 + kk + 16);
        v16h bv0 = __builtin_shufflevector(b0lo, b0hi, 0,1,2,3,4,5,6,7,8,9,10,11,12,13,14,15);
        v16h bv1 = __builtin_shufflevector(b1lo, b1hi, 0,1,2,3,4,5,6,7,8,9,10,11,12,13,14,15);

        acc0 = __builtin_amdgcn_wmma_f32_16x16x32_f16(false, a, false, bv0, (short)0, acc0, false, false);
        acc1 = __builtin_amdgcn_wmma_f32_16x16x32_f16(false, a, false, bv1, (short)0, acc1, false, false);
    }

    const int col0  = tn * 32 + l15;
    const int rbase = tm * 16 + ((lane & 16) ? 8 : 0);
    const float bv0s = bias ? bias[col0]      : 0.0f;
    const float bv1s = bias ? bias[col0 + 16] : 0.0f;
#pragma unroll
    for (int i = 0; i < 8; ++i) {
        float y0 = acc0[i] + bv0s;
        float y1 = acc1[i] + bv1s;
        if (ACT == 1) { y0 = silu_f(y0); y1 = silu_f(y1); }
        const size_t r = (size_t)(rbase + i) * ldc;
        if (OUT_F16) {
            ((_Float16*)Cout)[r + col0]      = (_Float16)y0;
            ((_Float16*)Cout)[r + col0 + 16] = (_Float16)y1;
        } else {
            ((float*)Cout)[r + col0]      = y0;
            ((float*)Cout)[r + col0 + 16] = y1;
        }
    }
}

// Weight prep: W (KxN f32, row-major) -> Wt (NxKp f16, zero-padded K)
__global__ void wprep_kernel(const float* __restrict__ W, _Float16* __restrict__ Wt,
                             int K, int N, int Kp)
{
    int idx = blockIdx.x * blockDim.x + threadIdx.x;
    if (idx >= N * Kp) return;
    int n = idx / Kp, k = idx % Kp;
    Wt[idx] = (k < K) ? (_Float16)W[(size_t)k * N + n] : (_Float16)0.0f;
}

// ------------------------------- small utility kernels ----------------------

__global__ void zero_int_kernel(int* p, int n)
{
    int i = blockIdx.x * blockDim.x + threadIdx.x;
    if (i < n) p[i] = 0;
}

__global__ void count_batch_kernel(const int* __restrict__ batch, int* cnt, int N)
{
    int n = blockIdx.x * blockDim.x + threadIdx.x;
    if (n < N) atomicAdd(&cnt[batch[n]], 1);
}

__global__ void scan_small_kernel(const int* __restrict__ cnt, int* start, int B)
{
    if (threadIdx.x == 0 && blockIdx.x == 0) {
        int run = 0;
        for (int b = 0; b < B; ++b) { start[b] = run; run += cnt[b]; }
        start[B] = run;
    }
}

__global__ void posenc_kernel(const int* __restrict__ batch,
                              const int* __restrict__ gstart,
                              const int* __restrict__ gcnt,
                              float* __restrict__ enc, int N)
{
    int idx = blockIdx.x * blockDim.x + threadIdx.x;
    if (idx >= N * HID) return;
    int n = idx >> 7, i = idx & 127;
    int b = batch[n];
    float pos  = (float)(n - gstart[b]) / ((float)gcnt[b] - 1.0f);
    float rate = __powf(10000.0f, -(float)(2 * (i >> 1)) / 128.0f);
    float ang  = pos * rate;
    enc[idx] = (i & 1) ? __cosf(ang) : __sinf(ang);
}

__global__ void tfeat_kernel(const float* __restrict__ t, const float* __restrict__ gfpW,
                             const float* __restrict__ teW, const float* __restrict__ teb,
                             float* __restrict__ tfeat)
{
    __shared__ float sc[HID];
    int b = blockIdx.x, c = threadIdx.x;
    if (c < 64) {
        float xp = 6.283185307179586f * t[b] * gfpW[c];
        sc[c]      = __sinf(xp);
        sc[c + 64] = __cosf(xp);
    }
    __syncthreads();
    float s = teb[c];
    for (int k = 0; k < HID; ++k) s += sc[k] * teW[k * HID + c];
    tfeat[b * HID + c] = silu_f(s);
}

__global__ void tproj_kernel(const float* __restrict__ tfeat, const float* __restrict__ tepW,
                             const float* __restrict__ tepb, float* __restrict__ tprojg, int B)
{
    int b = threadIdx.x >> 4, j = threadIdx.x & 15;
    if (b >= B) return;
    float s = tepb[j];
    for (int k = 0; k < HID; ++k) s += tfeat[b * HID + k] * tepW[k * 16 + j];
    tprojg[b * 16 + j] = s;
}

__global__ void small_gemm_kernel(const float* __restrict__ A, const float* __restrict__ W,
                                  const float* __restrict__ b, float* __restrict__ C)
{
    int r = blockIdx.x, c = threadIdx.x;
    float s = b ? b[c] : 0.0f;
    for (int k = 0; k < HID; ++k) s += A[r * HID + k] * W[k * HID + c];
    C[r * HID + c] = s;
}

__global__ void h0_kernel(const float* __restrict__ coords, const float* __restrict__ inW,
                          const float* __restrict__ inb, float* __restrict__ h, int N)
{
    int idx = blockIdx.x * blockDim.x + threadIdx.x;
    if (idx >= N * HID) return;
    int n = idx >> 7, c = idx & 127;
    float s = inb[c];
#pragma unroll
    for (int j = 0; j < 3; ++j) s += coords[n * 3 + j] * inW[j * HID + c];
    h[idx] = s;
}

// edge attributes, padded to 64 features (51 real + 13 zeros)
__global__ void ea_kernel(const float* __restrict__ coords, const int* __restrict__ ei,
                          const int* __restrict__ batch, const float* __restrict__ tprojg,
                          float* __restrict__ ea, int E)
{
    int j = blockIdx.x * blockDim.x + threadIdx.x;
    if (j >= E) return;
    int s = ei[j], d = ei[E + j];
    float ev0 = coords[s * 3 + 0] - coords[d * 3 + 0];
    float ev1 = coords[s * 3 + 1] - coords[d * 3 + 1];
    float ev2 = coords[s * 3 + 2] - coords[d * 3 + 2];
    float el  = sqrtf(ev0 * ev0 + ev1 * ev1 + ev2 * ev2);
    float inv = 1.0f / fmaxf(el, 1e-12f);
    float e0 = ev0 * inv, e1 = ev1 * inv, e2 = ev2 * inv;
    const float stepv = 5.0f / 33.0f;   // RAD/(NB+1)
    float* fwd = ea + (size_t)j * 64;
    float* rev = ea + (size_t)(E + j) * 64;
#pragma unroll
    for (int bn = 0; bn < 32; ++bn) {
        float vv = stepv * (float)(bn + 1);
        float zq = (el - vv) / stepv;
        float g  = __expf(-zq * zq) / 1.12f;
        fwd[bn] = g; rev[bn] = g;
    }
    fwd[32] = e0;  fwd[33] = e1;  fwd[34] = e2;
    rev[32] = -e0; rev[33] = -e1; rev[34] = -e2;
    int gb = batch[s];
#pragma unroll
    for (int jj = 0; jj < 16; ++jj) {
        float tv = tprojg[gb * 16 + jj];
        fwd[35 + jj] = tv; rev[35 + jj] = tv;
    }
#pragma unroll
    for (int jj = 51; jj < 64; ++jj) { fwd[jj] = 0.0f; rev[jj] = 0.0f; }
}

__global__ void edges_build_kernel(const int* __restrict__ ei, int* __restrict__ sarr,
                                   int* deg, int E)
{
    int j = blockIdx.x * blockDim.x + threadIdx.x;
    if (j >= 2 * E) return;
    int s, d;
    if (j < E) { s = ei[j];     d = ei[E + j]; }
    else       { s = ei[j];     d = ei[j - E]; }   // j in [E,2E): ei[j] == ei[E + (j-E)]
    sarr[j] = s;
    atomicAdd(&deg[d], 1);
}

__global__ void scan_deg_kernel(const int* __restrict__ deg, int* __restrict__ starts, int N)
{
    __shared__ int part[256];
    int chunk = (N + 255) / 256;
    int lo = threadIdx.x * chunk;
    int hi = lo + chunk; if (hi > N) hi = N;
    int s = 0;
    for (int i = lo; i < hi; ++i) s += deg[i];
    part[threadIdx.x] = s;
    __syncthreads();
    if (threadIdx.x == 0) {
        int run = 0;
        for (int i = 0; i < 256; ++i) { int tv = part[i]; part[i] = run; run += tv; }
        starts[N] = run;
    }
    __syncthreads();
    int run = part[threadIdx.x];
    for (int i = lo; i < hi; ++i) { starts[i] = run; run += deg[i]; }
}

__global__ void scatter_edges_kernel(const int* __restrict__ ei, const int* __restrict__ starts,
                                     int* cursor, int* __restrict__ inlist, int E)
{
    int j = blockIdx.x * blockDim.x + threadIdx.x;
    if (j >= 2 * E) return;
    int d = (j < E) ? ei[E + j] : ei[j - E];
    int pos = atomicAdd(&cursor[d], 1);
    inlist[starts[d] + pos] = j;
}

__global__ void pack_x_kernel(const float* __restrict__ h, const float* __restrict__ tg,
                              const int* __restrict__ batch, float* __restrict__ x, int N)
{
    int idx = blockIdx.x * blockDim.x + threadIdx.x;
    if (idx >= N * HID) return;
    int n = idx >> 7, c = idx & 127;
    x[(size_t)n * 384 + c]       = h[idx];
    x[(size_t)n * 384 + 256 + c] = tg[batch[n] * HID + c];
}

// two-pass segment-softmax attention: one block (128 thr) per destination node
__global__ void attn_kernel(const float* __restrict__ q, const float* __restrict__ k,
                            const float* __restrict__ v, const _Float16* __restrict__ e,
                            const int* __restrict__ starts, const int* __restrict__ inlist,
                            const int* __restrict__ sarr, float* __restrict__ out)
{
    const int d   = blockIdx.x;
    const int tid = threadIdx.x;
    const float qd = q[(size_t)d * HID + tid];
    const int beg = starts[d], end = starts[d + 1];
    const float inv_sqrt = 0.25f;   // 1/sqrt(CH=16)

    float m = -3.4e38f;
    for (int p = beg; p < end; ++p) {
        int eid = inlist[p];
        int s   = sarr[eid];
        float kj = k[(size_t)s * HID + tid] + (float)e[(size_t)eid * HID + tid];
        float pr = qd * kj;
        pr += __shfl_xor(pr, 1, 32);
        pr += __shfl_xor(pr, 2, 32);
        pr += __shfl_xor(pr, 4, 32);
        pr += __shfl_xor(pr, 8, 32);
        m = fmaxf(m, pr * inv_sqrt);
    }
    float sw = 0.0f, acc = 0.0f;
    for (int p = beg; p < end; ++p) {
        int eid = inlist[p];
        int s   = sarr[eid];
        float evv = (float)e[(size_t)eid * HID + tid];
        float kj  = k[(size_t)s * HID + tid] + evv;
        float pr  = qd * kj;
        pr += __shfl_xor(pr, 1, 32);
        pr += __shfl_xor(pr, 2, 32);
        pr += __shfl_xor(pr, 4, 32);
        pr += __shfl_xor(pr, 8, 32);
        float w = __expf(pr * inv_sqrt - m);
        sw  += w;
        acc += (v[(size_t)s * HID + tid] + evv) * w;
    }
    out[(size_t)d * HID + tid] = acc / sw;
}

__global__ void beta_hc_kernel(const float* __restrict__ aout, const float* __restrict__ xr,
                               const float* __restrict__ bw, float* __restrict__ hc)
{
    __shared__ float red[HID];
    int n = blockIdx.x, c = threadIdx.x;
    float o = aout[(size_t)n * HID + c];
    float x = xr[(size_t)n * HID + c];
    red[c] = o * bw[c] + x * bw[HID + c] + (o - x) * bw[2 * HID + c];
    __syncthreads();
    for (int s = 64; s > 0; s >>= 1) {
        if (c < s) red[c] += red[c + s];
        __syncthreads();
    }
    float beta = 1.0f / (1.0f + __expf(-red[0]));
    hc[(size_t)n * HID + c] = beta * x + (1.0f - beta) * o;
}

__global__ void gn_mean_kernel(const float* __restrict__ hc, const int* __restrict__ gstart,
                               float* __restrict__ mean)
{
    int b = blockIdx.x >> 7, c = blockIdx.x & 127;
    int beg = gstart[b], end = gstart[b + 1];
    __shared__ float red[256];
    float s = 0.0f;
    for (int n = beg + threadIdx.x; n < end; n += 256) s += hc[(size_t)n * HID + c];
    red[threadIdx.x] = s;
    __syncthreads();
    for (int st = 128; st > 0; st >>= 1) {
        if (threadIdx.x < st) red[threadIdx.x] += red[threadIdx.x + st];
        __syncthreads();
    }
    if (threadIdx.x == 0) mean[b * HID + c] = red[0] / (float)(end - beg);
}

__global__ void gn_var_kernel(const float* __restrict__ hc, const int* __restrict__ gstart,
                              const float* __restrict__ mean, const float* __restrict__ ms,
                              float* __restrict__ var)
{
    int b = blockIdx.x >> 7, c = blockIdx.x & 127;
    int beg = gstart[b], end = gstart[b + 1];
    float mv = mean[b * HID + c] * ms[c];
    __shared__ float red[256];
    float s = 0.0f;
    for (int n = beg + threadIdx.x; n < end; n += 256) {
        float o = hc[(size_t)n * HID + c] - mv;
        s += o * o;
    }
    red[threadIdx.x] = s;
    __syncthreads();
    for (int st = 128; st > 0; st >>= 1) {
        if (threadIdx.x < st) red[threadIdx.x] += red[threadIdx.x + st];
        __syncthreads();
    }
    if (threadIdx.x == 0) var[b * HID + c] = red[0] / (float)(end - beg);
}

__global__ void gn_apply_kernel(float* __restrict__ h, const float* __restrict__ hc,
                                const float* __restrict__ mean, const float* __restrict__ var,
                                const float* __restrict__ ms, const float* __restrict__ w,
                                const float* __restrict__ bb, const int* __restrict__ batch, int N)
{
    int idx = blockIdx.x * blockDim.x + threadIdx.x;
    if (idx >= N * HID) return;
    int n = idx >> 7, c = idx & 127;
    int b = batch[n];
    float o  = hc[idx] - mean[b * HID + c] * ms[c];
    float hn = w[c] * o * rsqrtf(var[b * HID + c] + 1e-5f) + bb[c];
    h[idx] = silu_f(hn + h[idx]);   // h holds the residual
}

__global__ void out_kernel(const float* __restrict__ g1, const float* __restrict__ o2W,
                           const float* __restrict__ o2b, float* __restrict__ out, int N)
{
    int n = blockIdx.x * blockDim.x + threadIdx.x;
    if (n >= N) return;
    float s0 = o2b[0], s1 = o2b[1], s2 = o2b[2];
    const float* g = g1 + (size_t)n * HID;
    for (int c = 0; c < HID; ++c) {
        float gv = g[c];
        s0 += gv * o2W[c * 3 + 0];
        s1 += gv * o2W[c * 3 + 1];
        s2 += gv * o2W[c * 3 + 2];
    }
    out[n * 3 + 0] = s0; out[n * 3 + 1] = s1; out[n * 3 + 2] = s2;
}

// ---------------------------------------------------------------------------

extern "C" void kernel_launch(void* const* d_in, const int* in_sizes, int n_in,
                              void* d_out, int out_size, void* d_ws, size_t ws_size,
                              hipStream_t stream)
{
    (void)n_in; (void)out_size; (void)ws_size;
    const float* coords = (const float*)d_in[0];
    const float* t      = (const float*)d_in[1];
    const float* gfpW   = (const float*)d_in[2];
    const float* peW    = (const float*)d_in[3];
    const float* peb    = (const float*)d_in[4];
    const float* teW    = (const float*)d_in[5];
    const float* teb    = (const float*)d_in[6];
    const float* tepW   = (const float*)d_in[7];
    const float* tepb   = (const float*)d_in[8];
    const float* posW   = (const float*)d_in[9];    // 5x128x128
    const float* posb   = (const float*)d_in[10];
    const float* tW     = (const float*)d_in[11];
    const float* tb     = (const float*)d_in[12];
    const float* qW     = (const float*)d_in[13];   // 5x384x128
    const float* qb     = (const float*)d_in[14];
    const float* kW     = (const float*)d_in[15];
    const float* kb     = (const float*)d_in[16];
    const float* vW     = (const float*)d_in[17];
    const float* vb     = (const float*)d_in[18];
    const float* skW    = (const float*)d_in[19];
    const float* skb    = (const float*)d_in[20];
    const float* eW     = (const float*)d_in[21];   // 5x51x128
    const float* betaW  = (const float*)d_in[22];   // 5x384x1
    const float* nw     = (const float*)d_in[23];
    const float* nbp    = (const float*)d_in[24];
    const float* nms    = (const float*)d_in[25];
    const float* inW    = (const float*)d_in[26];
    const float* inb    = (const float*)d_in[27];
    const float* o1W    = (const float*)d_in[28];
    const float* o1b    = (const float*)d_in[29];
    const float* o2W    = (const float*)d_in[30];
    const float* o2b    = (const float*)d_in[31];
    const int*   batch  = (const int*)d_in[32];
    const int*   ei     = (const int*)d_in[33];

    const int N  = in_sizes[0] / 3;
    const int B  = in_sizes[1];
    const int E  = in_sizes[33] / 2;
    const int E2 = 2 * E;

    // scratch arena
    char* wp = (char*)d_ws;
    auto alloc = [&](size_t bytes) -> void* {
        void* p = wp; wp += (bytes + 255) & ~(size_t)255; return p;
    };
    float* pos_feat = (float*)alloc((size_t)N * HID * 4);
    float* enc      = (float*)alloc((size_t)N * HID * 4);
    float* h        = (float*)alloc((size_t)N * HID * 4);
    float* x        = (float*)alloc((size_t)N * 384 * 4);
    float* qbuf     = (float*)alloc((size_t)N * HID * 4);
    float* kbuf     = (float*)alloc((size_t)N * HID * 4);
    float* vbuf     = (float*)alloc((size_t)N * HID * 4);
    float* xrbuf    = (float*)alloc((size_t)N * HID * 4);
    float* attn     = (float*)alloc((size_t)N * HID * 4);
    float* hc       = (float*)alloc((size_t)N * HID * 4);
    float* g1       = (float*)alloc((size_t)N * HID * 4);
    float* ea       = (float*)alloc((size_t)E2 * 64 * 4);          // padded to 64
    _Float16* ef16  = (_Float16*)alloc((size_t)E2 * HID * 2);
    float* tfeat    = (float*)alloc((size_t)B * HID * 4);
    float* tprojg   = (float*)alloc((size_t)B * 16 * 4);
    float* tg       = (float*)alloc((size_t)B * HID * 4);
    float* meanb    = (float*)alloc((size_t)B * HID * 4);
    float* varb     = (float*)alloc((size_t)B * HID * 4);
    int* gcnt   = (int*)alloc((size_t)B * 4);
    int* gstart = (int*)alloc((size_t)(B + 1) * 4);
    int* deg    = (int*)alloc((size_t)N * 4);
    int* starts = (int*)alloc((size_t)(N + 1) * 4);
    int* cursor = (int*)alloc((size_t)N * 4);
    int* sarr   = (int*)alloc((size_t)E2 * 4);
    int* inlist = (int*)alloc((size_t)E2 * 4);
    // f16 transposed weights
    _Float16* peWt  = (_Float16*)alloc((size_t)HID * HID * 2);
    _Float16* posWt = (_Float16*)alloc((size_t)NL * HID * HID * 2);
    _Float16* qWt   = (_Float16*)alloc((size_t)NL * HID * 384 * 2);
    _Float16* kWt   = (_Float16*)alloc((size_t)NL * HID * 384 * 2);
    _Float16* vWt   = (_Float16*)alloc((size_t)NL * HID * 384 * 2);
    _Float16* skWt  = (_Float16*)alloc((size_t)NL * HID * 384 * 2);
    _Float16* eWt   = (_Float16*)alloc((size_t)NL * HID * 64 * 2); // Kp=64
    _Float16* o1Wt  = (_Float16*)alloc((size_t)HID * HID * 2);

    const int NC = N * HID;
    const dim3 blk256(256);
    const dim3 gNC((NC + 255) / 256);

    auto wprep = [&](const float* W, _Float16* Wt, int K, int Nn, int Kp) {
        int tot = Nn * Kp;
        wprep_kernel<<<dim3((tot + 255) / 256), blk256, 0, stream>>>(W, Wt, K, Nn, Kp);
    };
    auto gemm = [&](const float* A, int lda, const _Float16* Bt, int ldb,
                    const float* bias, void* C, int ldc, int M, int Nn, int K,
                    int act, bool outF16) {
        int tiles = (M / 16) * (Nn / 32);
        dim3 blk(128), grd((tiles + 3) / 4);
        if (outF16) {
            if (act == 1) wmma_gemm<1, true><<<grd, blk, 0, stream>>>(A, lda, Bt, ldb, bias, C, ldc, M, Nn, K);
            else          wmma_gemm<0, true><<<grd, blk, 0, stream>>>(A, lda, Bt, ldb, bias, C, ldc, M, Nn, K);
        } else {
            if (act == 1) wmma_gemm<1, false><<<grd, blk, 0, stream>>>(A, lda, Bt, ldb, bias, C, ldc, M, Nn, K);
            else          wmma_gemm<0, false><<<grd, blk, 0, stream>>>(A, lda, Bt, ldb, bias, C, ldc, M, Nn, K);
        }
    };

    // ---- weight prep (f16 transpose, zero-padded K) ----
    wprep(peW, peWt, HID, HID, HID);
    wprep(o1W, o1Wt, HID, HID, HID);
    for (int l = 0; l < NL; ++l) {
        wprep(posW + (size_t)l * HID * HID, posWt + (size_t)l * HID * HID, HID, HID, HID);
        wprep(qW  + (size_t)l * 384 * HID, qWt  + (size_t)l * HID * 384, 384, HID, 384);
        wprep(kW  + (size_t)l * 384 * HID, kWt  + (size_t)l * HID * 384, 384, HID, 384);
        wprep(vW  + (size_t)l * 384 * HID, vWt  + (size_t)l * HID * 384, 384, HID, 384);
        wprep(skW + (size_t)l * 384 * HID, skWt + (size_t)l * HID * 384, 384, HID, 384);
        wprep(eW  + (size_t)l * 51 * HID,  eWt  + (size_t)l * HID * 64,  51,  HID, 64);
    }

    // ---- graph segment bookkeeping ----
    zero_int_kernel<<<dim3((B + 255) / 256), blk256, 0, stream>>>(gcnt, B);
    count_batch_kernel<<<dim3((N + 255) / 256), blk256, 0, stream>>>(batch, gcnt, N);
    scan_small_kernel<<<dim3(1), dim3(32), 0, stream>>>(gcnt, gstart, B);

    // ---- positional / time features ----
    posenc_kernel<<<gNC, blk256, 0, stream>>>(batch, gstart, gcnt, enc, N);
    gemm(enc, HID, peWt, HID, peb, pos_feat, HID, N, HID, HID, /*silu*/1, false);
    tfeat_kernel<<<dim3(B), dim3(HID), 0, stream>>>(t, gfpW, teW, teb, tfeat);
    tproj_kernel<<<dim3(1), dim3(128), 0, stream>>>(tfeat, tepW, tepb, tprojg, B);

    // ---- node embedding ----
    h0_kernel<<<gNC, blk256, 0, stream>>>(coords, inW, inb, h, N);

    // ---- incoming-edge CSR over concatenated edge list ----
    zero_int_kernel<<<dim3((N + 255) / 256), blk256, 0, stream>>>(deg, N);
    edges_build_kernel<<<dim3((E2 + 255) / 256), blk256, 0, stream>>>(ei, sarr, deg, E);
    scan_deg_kernel<<<dim3(1), blk256, 0, stream>>>(deg, starts, N);
    zero_int_kernel<<<dim3((N + 255) / 256), blk256, 0, stream>>>(cursor, N);
    scatter_edges_kernel<<<dim3((E2 + 255) / 256), blk256, 0, stream>>>(ei, starts, cursor, inlist, E);

    // ---- edge attributes (fwd + rev, 64-padded) ----
    ea_kernel<<<dim3((E + 255) / 256), blk256, 0, stream>>>(coords, ei, batch, tprojg, ea, E);

    // ---- transformer layers ----
    for (int l = 0; l < NL; ++l) {
        const float* posbl = posb + (size_t)l * HID;
        const float* tWl   = tW   + (size_t)l * HID * HID;
        const float* tbl   = tb   + (size_t)l * HID;
        const float* qbl   = qb   + (size_t)l * HID;
        const float* kbl   = kb   + (size_t)l * HID;
        const float* vbl   = vb   + (size_t)l * HID;
        const float* skbl  = skb  + (size_t)l * HID;
        const float* bWl   = betaW + (size_t)l * 384;

        small_gemm_kernel<<<dim3(B), dim3(HID), 0, stream>>>(tfeat, tWl, tbl, tg);
        pack_x_kernel<<<gNC, blk256, 0, stream>>>(h, tg, batch, x, N);
        gemm(pos_feat, HID, posWt + (size_t)l * HID * HID, HID, posbl, x + HID, 384, N, HID, HID, 0, false);

        gemm(x, 384, qWt  + (size_t)l * HID * 384, 384, qbl,  qbuf,  HID, N, HID, 384, 0, false);
        gemm(x, 384, kWt  + (size_t)l * HID * 384, 384, kbl,  kbuf,  HID, N, HID, 384, 0, false);
        gemm(x, 384, vWt  + (size_t)l * HID * 384, 384, vbl,  vbuf,  HID, N, HID, 384, 0, false);
        gemm(x, 384, skWt + (size_t)l * HID * 384, 384, skbl, xrbuf, HID, N, HID, 384, 0, false);

        gemm(ea, 64, eWt + (size_t)l * HID * 64, 64, nullptr, ef16, HID, E2, HID, 64, 0, true);

        attn_kernel<<<dim3(N), dim3(HID), 0, stream>>>(qbuf, kbuf, vbuf, ef16,
                                                       starts, inlist, sarr, attn);
        beta_hc_kernel<<<dim3(N), dim3(HID), 0, stream>>>(attn, xrbuf, bWl, hc);

        gn_mean_kernel<<<dim3(B * HID), blk256, 0, stream>>>(hc, gstart, meanb);
        gn_var_kernel<<<dim3(B * HID), blk256, 0, stream>>>(hc, gstart, meanb,
                                                            nms + (size_t)l * HID, varb);
        gn_apply_kernel<<<gNC, blk256, 0, stream>>>(h, hc, meanb, varb,
                                                    nms + (size_t)l * HID,
                                                    nw + (size_t)l * HID,
                                                    nbp + (size_t)l * HID, batch, N);
    }

    // ---- output head ----
    gemm(h, HID, o1Wt, HID, o1b, g1, HID, N, HID, HID, /*silu*/1, false);
    out_kernel<<<dim3((N + 255) / 256), blk256, 0, stream>>>(g1, o2W, o2b, (float*)d_out, N);
}
